// WaveEncoder_41343355191625
// MI455X (gfx1250) — compile-verified, matrix operand order checked by reading, not run on hardware
//
#include <hip/hip_runtime.h>
#include <hip/hip_bf16.h>

typedef __attribute__((ext_vector_type(16))) __bf16 v16bf;
typedef __attribute__((ext_vector_type(8)))  float  v8f;

#define T  4096
#define B  8
#define R  64
#define SL ((size_t)B * R * T)   // elements per state slab [B,64,T]

__device__ __forceinline__ float fsig(float x) {
  return 1.0f / (1.0f + __expf(-x));
}
__device__ __forceinline__ float ftanh(float x) {
#if __has_builtin(__builtin_amdgcn_tanhf)
  return __builtin_amdgcn_tanhf(x);
#elif __has_builtin(__builtin_amdgcn_tanh_f32)
  return __builtin_amdgcn_tanh_f32(x);
#else
  x = fminf(fmaxf(x, -15.0f), 15.0f);
  float e = __expf(2.0f * x);
  return (e - 1.0f) / (e + 1.0f);
#endif
}

// ---------------- init: h0 = tanh(Wh @ in + bh), c0 = tanh(Wc @ in + bc) ----
__global__ __launch_bounds__(256) void wave_init(
    const float* __restrict__ x, const float* __restrict__ feat,
    const float* __restrict__ whp, const float* __restrict__ bhp,
    const float* __restrict__ wcp, const float* __restrict__ bcp,
    float* __restrict__ h0, float* __restrict__ Cws)
{
  __shared__ float wh[512], wc[512], bh[64], bc[64];
  const int tid = threadIdx.x;
  for (int k = tid; k < 512; k += 256) { wh[k] = whp[k]; wc[k] = wcp[k]; }
  if (tid < 64) { bh[tid] = bhp[tid]; bc[tid] = bcp[tid]; }
  __syncthreads();

  const int g = blockIdx.x * 256 + tid;      // over B*T
  const int b = g >> 12, t = g & (T - 1);
  float in[8];
  in[0] = x[(size_t)b * T + t];
#pragma unroll
  for (int f = 0; f < 7; ++f) in[1 + f] = feat[((size_t)b * 7 + f) * T + t];

  for (int o = 0; o < R; ++o) {
    float sh = bh[o], sc = bc[o];
#pragma unroll
    for (int i = 0; i < 8; ++i) {
      sh = fmaf(wh[o * 8 + i], in[i], sh);
      sc = fmaf(wc[o * 8 + i], in[i], sc);
    }
    const size_t ix = ((size_t)b * R + o) * T + t;
    h0[ix]  = ftanh(sh);
    Cws[ix] = ftanh(sc);
  }
}

// ---------------- one gated dilated-conv layer: WMMA GEMM + gating ----------
//
// LDS holds WMMA operands in *fragment-major* layout: fragment f is 32
// contiguous bytes per lane at  base + f*1024 + lane*32, so an operand load
// is a single 32B LDS read (2x ds_load_b128, immediate offsets, no VALU).
//
// A: 16 M-tiles x 4 K-steps = 64 frags = 64 KB (all offsets < 65536).
// B:  8 waves   x 4 K-steps = 32 frags = 32 KB.
//
// K-mapping inside a fragment (16-bit operands, wave32), with r = k & 31:
//   hi = (r>>3)&1,  j = (r&7) | ((r>>4)<<3)   (inverse of ISA table)
__global__ __launch_bounds__(256) void wave_layer(
    const float* __restrict__ hin, float* __restrict__ hout,
    float* __restrict__ Cws, const float* __restrict__ w,
    const float* __restrict__ bias, int d)
{
  __shared__ __bf16 ldsA[16 * 4 * 32 * 16];   // 64 KB
  __shared__ __bf16 ldsB[ 8 * 4 * 32 * 16];   // 32 KB
  __shared__ float  lds_bias[256];

  const int tid = threadIdx.x;
  const int b   = blockIdx.y;
  const int t0  = blockIdx.x * 128;

  // ---- stage weights A[o][k] (k<64: tap t-d, else tap t), swizzled ----
  for (int idx = tid; idx < 256 * 128; idx += 256) {
    const int o = idx >> 7, k = idx & 127;
    const float v = (k < 64) ? w[(o * 64 + k) * 2]
                             : w[(o * 64 + (k - 64)) * 2 + 1];
    const int m = o >> 4, lo = o & 15;
    const int s = k >> 5, r = k & 31;
    const int hi = (r >> 3) & 1;
    const int j  = (r & 7) | ((r >> 4) << 3);
    ldsA[(((m * 4 + s) * 32) + hi * 16 + lo) * 16 + j] = (__bf16)v;
  }
  if (tid < 256) lds_bias[tid] = bias[tid];

  // ---- stage activations B[k][t] (k<64: h(t-d), else h(t)), swizzled ----
  const float* hb = hin + (size_t)b * R * T;
  for (int idx = tid; idx < 128 * 128; idx += 256) {
    const int k = idx >> 7, tl = idx & 127;      // consecutive tid -> t: coalesced
    const int t = t0 + tl - ((k < 64) ? d : 0);
    const float v = (t >= 0) ? hb[(size_t)(k & 63) * T + t] : 0.0f;
    const int s = k >> 5, r = k & 31;
    const int hi = (r >> 3) & 1;
    const int j  = (r & 7) | ((r >> 4) << 3);
    const int wv = tl >> 4, lo = tl & 15;
    ldsB[(((wv * 4 + s) * 32) + hi * 16 + lo) * 16 + j] = (__bf16)v;
  }
  __syncthreads();

  const int wave = tid >> 5, lane = tid & 31;
  const int n = lane & 15, hi = lane >> 4;

  // B fragments for this wave's 16 time columns (loaded once)
  v16bf bf[4];
#pragma unroll
  for (int s = 0; s < 4; ++s)
    bf[s] = *(const v16bf*)(ldsB + ((wave * 4 + s) * 32 + lane) * 16);

  const int t = t0 + wave * 16 + n;
  float* Cb = Cws  + (size_t)b * R * T;
  float* Hb = hout + (size_t)b * R * T;
  const __bf16* aLane = ldsA + lane * 16;

  for (int mq = 0; mq < 4; ++mq) {        // 16-row group within each gate
    v8f acc[4];
#pragma unroll
    for (int q = 0; q < 4; ++q) acc[q] = (v8f)0.0f;

#pragma unroll
    for (int q = 0; q < 4; ++q) {         // gate quarter: ig, cf, cg, eg
      const int m = q * 4 + mq;
#pragma unroll
      for (int s = 0; s < 4; ++s) {
        const v16bf af = *(const v16bf*)(aLane + (m * 4 + s) * 512);
        acc[q] = __builtin_amdgcn_wmma_f32_16x16x32_bf16(
            false, af, false, bf[s], (short)0, acc[q], false, false);
      }
    }

    // fused gating: c' = sig(ig)*c + tanh(cf)*sig(cg); h' = sig(eg)*tanh(c')
#pragma unroll
    for (int r = 0; r < 8; ++r) {
      const int row = mq * 16 + (hi ? (r + 8) : r);   // channel 0..63
      const float ig = acc[0][r] + lds_bias[row];
      const float cf = acc[1][r] + lds_bias[64 + row];
      const float cg = acc[2][r] + lds_bias[128 + row];
      const float eg = acc[3][r] + lds_bias[192 + row];
      const size_t ix = (size_t)row * T + t;
      const float cold = Cb[ix];
      const float cnew = fsig(ig) * cold + ftanh(cf) * fsig(cg);
      Cb[ix] = cnew;
      Hb[ix] = fsig(eg) * ftanh(cnew);
    }
  }
}

// ---------------------------------------------------------------------------
extern "C" void kernel_launch(void* const* d_in, const int* in_sizes, int n_in,
                              void* d_out, int out_size, void* d_ws, size_t ws_size,
                              hipStream_t stream) {
  (void)in_sizes; (void)n_in; (void)out_size; (void)ws_size;
  const float* x      = (const float*)d_in[0];
  const float* feat   = (const float*)d_in[1];
  const float* fc_h_w = (const float*)d_in[2];
  const float* fc_h_b = (const float*)d_in[3];
  const float* fc_c_w = (const float*)d_in[4];
  const float* fc_c_b = (const float*)d_in[5];
  const float* conv_w = (const float*)d_in[6];
  const float* conv_b = (const float*)d_in[7];
  float* out = (float*)d_out;
  float* Cws = (float*)d_ws;               // 8 MB: running cell state [B,64,T]

  wave_init<<<(B * T) / 256, 256, 0, stream>>>(
      x, feat, fc_h_w, fc_h_b, fc_c_w, fc_c_b, out, Cws);

  for (int i = 0; i < 23; ++i) {
    wave_layer<<<dim3(T / 128, B), 256, 0, stream>>>(
        out + (size_t)i * SL, out + (size_t)(i + 1) * SL, Cws,
        conv_w + (size_t)i * 256 * 64 * 2, conv_b + (size_t)i * 256,
        1 << (i & 7));
  }
}
